// SELF_ATTENTION_11940009083330
// MI455X (gfx1250) — matrix operand order repaired and checked by reading, not compiled, not run on hardware
//
#include <hip/hip_runtime.h>
#include <hip/hip_bf16.h>

typedef __attribute__((ext_vector_type(16))) _Float16 v16h;
typedef __attribute__((ext_vector_type(8)))  _Float16 v8h;
typedef __attribute__((ext_vector_type(4)))  _Float16 v4h;
typedef __attribute__((ext_vector_type(8)))  float    v8f;

#define HQ_   16
#define HKV_  4
#define HD_   128
#define SEQ_  2048
#define DMODEL_ 2048
#define BATCH_ 2

#define USE_ASYNC_LDS 1   // gfx1250 global_load_async_to_lds_b128 staging

// ---------------------------------------------------------------------------
// WMMA helpers (CDNA5 / gfx1250, wave32)
// ---------------------------------------------------------------------------
__device__ __forceinline__ v8f wmma_f16(v16h a, v16h b, v8f c) {
  // D = A(16x32 f16) * B(32x16 f16) + C(16x16 f32)
  return __builtin_amdgcn_wmma_f32_16x16x32_f16(
      /*neg_a=*/false, a, /*neg_b=*/false, b,
      /*c_mod=*/(short)0, c, /*reuse_a=*/false, /*reuse_b=*/false);
}

// A fragment (16x32 f16, row-major source, row stride ld halfs).
// ISA layout: lane<16 -> row=lane,   K = {0..7, 16..23}
//             lane>=16 -> row=lane-16, K = {8..15, 24..31}
__device__ __forceinline__ v16h load_a_frag(const _Float16* __restrict__ tile, int ld) {
  const int lane = threadIdx.x & 31;
  const int r  = lane & 15;
  const int kb = (lane >> 4) << 3;          // 0 or 8
  const _Float16* p = tile + (size_t)r * ld + kb;
  v8h lo = *(const v8h*)(p);
  v8h hi = *(const v8h*)(p + 16);
  return __builtin_shufflevector(lo, hi, 0,1,2,3,4,5,6,7,8,9,10,11,12,13,14,15);
}

// B fragment (32x16 f16) from column-major storage: tile[col*ld + k].
// ISA layout: lane<16 -> col=lane,   K = {0..7, 16..23}
//             lane>=16 -> col=lane-16, K = {8..15, 24..31}
__device__ __forceinline__ v16h load_b_frag(const _Float16* __restrict__ tile, int ld) {
  const int lane = threadIdx.x & 31;
  const int c  = lane & 15;
  const int kb = (lane >> 4) << 3;          // 0 or 8
  const _Float16* p = tile + (size_t)c * ld + kb;
  v8h lo = *(const v8h*)(p);
  v8h hi = *(const v8h*)(p + 16);
  return __builtin_shufflevector(lo, hi, 0,1,2,3,4,5,6,7,8,9,10,11,12,13,14,15);
}

#if USE_ASYNC_LDS
// gfx1250 async global -> LDS copy, 16 bytes per lane, tracked by ASYNCcnt.
// Generic shared pointers carry the LDS offset in their low 32 bits
// (aperture in [63:32]); hardware uses addr[31:0] as the LDS address.
__device__ __forceinline__ void async_copy_b128(const _Float16* gptr, _Float16* lptr) {
  unsigned lds = (unsigned)(unsigned long long)lptr;
  asm volatile("global_load_async_to_lds_b128 %0, %1, off"
               :: "v"(lds), "v"(gptr) : "memory");
}
__device__ __forceinline__ void wait_async() {
  asm volatile("s_wait_asynccnt 0x0" ::: "memory");
}
#endif

// ---------------------------------------------------------------------------
// f32 -> f16 cast (vectorized x4)
// ---------------------------------------------------------------------------
__global__ __launch_bounds__(256) void cast_f32_f16(const float* __restrict__ in,
                                                    _Float16* __restrict__ out, int n4) {
  int i = blockIdx.x * 256 + threadIdx.x;
  if (i < n4) {
    float4 f = ((const float4*)in)[i];
    v4h h = {(_Float16)f.x, (_Float16)f.y, (_Float16)f.z, (_Float16)f.w};
    ((v4h*)out)[i] = h;
  }
}

// ---------------------------------------------------------------------------
// f32 -> f16 cast + transpose: in [K][N] f32 -> out [N][K] f16.
// One-time cost for weights so GEMM B-tiles stage as contiguous b128 copies.
// ---------------------------------------------------------------------------
__global__ __launch_bounds__(256) void cast_t_f32_f16(const float* __restrict__ in,
                                                      _Float16* __restrict__ out,
                                                      int K, int N) {
  int i = blockIdx.x * 256 + threadIdx.x;
  if (i >= K * N) return;
  int n = i % N;                 // consecutive threads: consecutive n -> coalesced reads
  int k = i / N;
  out[(size_t)n * K + k] = (_Float16)in[i];
}

// ---------------------------------------------------------------------------
// WMMA GEMM: C[M,N] = A[M,K] * Bt[N,K]^T,  A row-major f16, Bt = B transposed.
// Block = 256 threads (8 waves), tile 128x128, BK=32. Wave tile = 32x64.
// Tile staging via gfx1250 async global->LDS b128 (ASYNCcnt).
// ---------------------------------------------------------------------------
template <typename CT>
__global__ __launch_bounds__(256) void wmma_gemm_kernel(
    const _Float16* __restrict__ A, const _Float16* __restrict__ Bt,
    CT* __restrict__ C, int M, int N, int K) {
  __shared__ _Float16 As[128 * 48];   // row-major,    ld=48 (32 + pad, 16B-aligned rows)
  __shared__ _Float16 Bs[128 * 48];   // column-major: Bs[col*48 + k]

  const int tid  = threadIdx.x;
  const int lane = tid & 31;
  const int w    = tid >> 5;          // 0..7
  const int wm   = w & 3;             // 4 wave-rows
  const int wn   = w >> 2;            // 2 wave-cols
  const int m0   = blockIdx.y * 128;
  const int n0   = blockIdx.x * 128;

  v8f zf = {};
  v8f acc[2][4];
#pragma unroll
  for (int i = 0; i < 2; ++i)
#pragma unroll
    for (int j = 0; j < 4; ++j) acc[i][j] = zf;

  for (int k0 = 0; k0 < K; k0 += 32) {
#if USE_ASYNC_LDS
    // A tile 128x32 and B tile 128(cols)x32(k): 2+2 async b128 per thread
#pragma unroll
    for (int i = tid; i < 512; i += 256) {
      int m  = i >> 2;
      int kc = (i & 3) << 3;
      async_copy_b128(A + (size_t)(m0 + m) * K + k0 + kc, As + m * 48 + kc);
    }
#pragma unroll
    for (int i = tid; i < 512; i += 256) {
      int n  = i >> 2;
      int kc = (i & 3) << 3;
      async_copy_b128(Bt + (size_t)(n0 + n) * K + k0 + kc, Bs + n * 48 + kc);
    }
    wait_async();
#else
#pragma unroll
    for (int i = tid; i < 512; i += 256) {
      int m  = i >> 2;
      int kc = (i & 3) << 3;
      *(v8h*)(As + m * 48 + kc) = *(const v8h*)(A + (size_t)(m0 + m) * K + k0 + kc);
    }
#pragma unroll
    for (int i = tid; i < 512; i += 256) {
      int n  = i >> 2;
      int kc = (i & 3) << 3;
      *(v8h*)(Bs + n * 48 + kc) = *(const v8h*)(Bt + (size_t)(n0 + n) * K + k0 + kc);
    }
#endif
    __syncthreads();

    if (k0 + 32 < K) {  // hint next tiles toward L2 (global_prefetch_b8)
      __builtin_prefetch(A + (size_t)(m0 + (tid >> 1)) * K + k0 + 32, 0, 1);
      __builtin_prefetch(Bt + (size_t)(n0 + (tid >> 1)) * K + k0 + 32, 0, 1);
    }

    v16h af[2], bf[4];
#pragma unroll
    for (int mf = 0; mf < 2; ++mf)
      af[mf] = load_a_frag(As + (wm * 32 + mf * 16) * 48, 48);
#pragma unroll
    for (int nf = 0; nf < 4; ++nf)
      bf[nf] = load_b_frag(Bs + (wn * 64 + nf * 16) * 48, 48);
#pragma unroll
    for (int mf = 0; mf < 2; ++mf)
#pragma unroll
      for (int nf = 0; nf < 4; ++nf)
        acc[mf][nf] = wmma_f16(af[mf], bf[nf], acc[mf][nf]);
    __syncthreads();
  }

  // --- store: C frag layout: element (M = j + 8*(lane>=16), N = lane&15) ---
  const int hi = lane >> 4;
  const int c  = lane & 15;
#pragma unroll
  for (int mf = 0; mf < 2; ++mf)
#pragma unroll
    for (int nf = 0; nf < 4; ++nf)
#pragma unroll
      for (int j = 0; j < 8; ++j) {
        int row = m0 + wm * 32 + mf * 16 + j + 8 * hi;
        int col = n0 + wn * 64 + nf * 16 + c;
        C[(size_t)row * N + col] = (CT)acc[mf][nf][j];
      }
}

// ---------------------------------------------------------------------------
// RoPE in-place on f16 [b, t, heads, 128]; one thread per pair.
// ---------------------------------------------------------------------------
__global__ __launch_bounds__(256) void rope_kernel(_Float16* __restrict__ d,
                                                   int hp /* heads*64 */, int npairs) {
  int p = blockIdx.x * 256 + threadIdx.x;
  if (p >= npairs) return;
  int d2 = p & 63;
  int t  = (p / hp) & (SEQ_ - 1);
  float freq = __powf(10000.0f, -(float)(2 * d2) * (1.0f / 128.0f));
  float sn, cs;
  __sincosf((float)t * freq, &sn, &cs);
  float x1 = (float)d[2 * p];
  float x2 = (float)d[2 * p + 1];
  d[2 * p]     = (_Float16)(x1 * cs - x2 * sn);
  d[2 * p + 1] = (_Float16)(x1 * sn + x2 * cs);
}

// ---------------------------------------------------------------------------
// V transpose: [b, t, kv, d] -> [b, kv, d, t]  (so PV B-fragments are contiguous)
// ---------------------------------------------------------------------------
__global__ __launch_bounds__(256) void vtrans_kernel(const _Float16* __restrict__ v,
                                                     _Float16* __restrict__ vt) {
  int i = blockIdx.x * 256 + threadIdx.x;
  if (i >= BATCH_ * SEQ_ * HKV_ * HD_) return;
  int dd = i & 127;
  int kv = (i >> 7) & 3;
  int t  = (i >> 9) & (SEQ_ - 1);
  int b  = i >> 20;
  vt[((size_t)(b * HKV_ + kv) * HD_ + dd) * SEQ_ + t] = v[i];
}

// ---------------------------------------------------------------------------
// Flash attention: one wave per 16-row Q tile, online softmax, 32-key blocks.
// q: [b,t,16,128] f16 (roped); k: [b,t,4,128] f16 (roped); vt: [b,kv,128,t] f16
// out ah: [b,t,16,128] f16
// ---------------------------------------------------------------------------
__global__ __launch_bounds__(128) void flash_attn_kernel(
    const _Float16* __restrict__ q, const _Float16* __restrict__ k,
    const _Float16* __restrict__ vt, _Float16* __restrict__ ah) {
  __shared__ _Float16 pbuf[4][16 * 32];   // per-wave P staging (C-layout -> A-layout)

  const int lane = threadIdx.x & 31;
  const int w    = threadIdx.x >> 5;
  const int idx  = blockIdx.x * 4 + w;    // 0 .. 4095
  const int qt   = idx & 127;             // 128 q-tiles per (b,h)
  const int h    = (idx >> 7) & 15;
  const int b    = idx >> 11;
  const int kv   = h >> 2;                // GQA: 4 q-heads per kv-head
  const int t0   = qt * 16;

  const int hi = lane >> 4;
  const int c  = lane & 15;
  const float scale = 0.08838834764831845f;  // 1/sqrt(128)

  // Q fragments: 16 rows x 128, as 4 K-chunks of 32
  v16h qf[4];
  const _Float16* qbase = q + ((size_t)(b * SEQ_ + t0) * HQ_ + h) * HD_;
#pragma unroll
  for (int kc = 0; kc < 4; ++kc)
    qf[kc] = load_a_frag(qbase + kc * 32, HQ_ * HD_);

  v8f zf = {};
  v8f o[8];
#pragma unroll
  for (int nt = 0; nt < 8; ++nt) o[nt] = zf;
  float mrow[8], lrow[8];
#pragma unroll
  for (int j = 0; j < 8; ++j) { mrow[j] = -1e30f; lrow[j] = 0.0f; }

  const _Float16* vbase = vt + ((size_t)(b * HKV_ + kv) * HD_) * SEQ_;
  const int nkb = ((t0 + 15) >> 5) + 1;   // causal: key blocks of 32

  for (int kb = 0; kb < nkb; ++kb) {
    const int kbase = kb * 32;

    // ---- S = Q * K^T for 32 keys (two 16x16 frags) ----
    v8f s0 = zf, s1 = zf;
    const _Float16* kb0 = k + ((size_t)(b * SEQ_ + kbase) * HKV_ + kv) * HD_;
    const _Float16* kb1 = kb0 + (size_t)16 * HKV_ * HD_;
#pragma unroll
    for (int kc = 0; kc < 4; ++kc) {
      v16h bf0 = load_b_frag(kb0 + kc * 32, HKV_ * HD_);
      v16h bf1 = load_b_frag(kb1 + kc * 32, HKV_ * HD_);
      s0 = wmma_f16(qf[kc], bf0, s0);
      s1 = wmma_f16(qf[kc], bf1, s1);
    }
    if (kb + 1 < nkb) {
      __builtin_prefetch(kb0 + (size_t)32 * HKV_ * HD_ + lane * 16, 0, 1);
      __builtin_prefetch(vbase + kbase + 32 + (size_t)lane * 4 * SEQ_, 0, 1);
    }

    // ---- causal mask + scale + per-row max (lane = column, vgpr = row) ----
    float pm[8];
#pragma unroll
    for (int j = 0; j < 8; ++j) {
      int trow = t0 + j + 8 * hi;
      float v0 = (kbase + c      <= trow) ? s0[j] * scale : -1e30f;
      float v1 = (kbase + 16 + c <= trow) ? s1[j] * scale : -1e30f;
      s0[j] = v0; s1[j] = v1;
      pm[j] = fmaxf(v0, v1);
    }
#pragma unroll
    for (int msk = 1; msk <= 8; msk <<= 1)
#pragma unroll
      for (int j = 0; j < 8; ++j)
        pm[j] = fmaxf(pm[j], __shfl_xor(pm[j], msk));

    // ---- online-softmax update ----
    float alpha[8], ps[8];
#pragma unroll
    for (int j = 0; j < 8; ++j) {
      float mnew = fmaxf(mrow[j], pm[j]);
      alpha[j] = __expf(mrow[j] - mnew);
      mrow[j] = mnew;
      float p0 = __expf(s0[j] - mnew);
      float p1 = __expf(s1[j] - mnew);
      s0[j] = p0; s1[j] = p1;
      ps[j] = p0 + p1;
    }
#pragma unroll
    for (int msk = 1; msk <= 8; msk <<= 1)
#pragma unroll
      for (int j = 0; j < 8; ++j)
        ps[j] += __shfl_xor(ps[j], msk);
#pragma unroll
    for (int j = 0; j < 8; ++j) lrow[j] = lrow[j] * alpha[j] + ps[j];
#pragma unroll
    for (int nt = 0; nt < 8; ++nt)
#pragma unroll
      for (int j = 0; j < 8; ++j) o[nt][j] *= alpha[j];

    // ---- P: C-layout -> A-layout through LDS (wave-internal, in-order) ----
    _Float16* pl = pbuf[w];
#pragma unroll
    for (int j = 0; j < 8; ++j) {
      int r = j + 8 * hi;
      pl[r * 32 + c]      = (_Float16)s0[j];
      pl[r * 32 + 16 + c] = (_Float16)s1[j];
    }
    __builtin_amdgcn_wave_barrier();
    v16h pf = load_a_frag(pl, 32);

    // ---- O += P * V (8 N-tiles of 16) ----
#pragma unroll
    for (int nt = 0; nt < 8; ++nt) {
      v16h vf = load_b_frag(vbase + (size_t)(nt * 16) * SEQ_ + kbase, SEQ_);
      o[nt] = wmma_f16(pf, vf, o[nt]);
    }
  }

  // ---- normalize + store ----
  float inv[8];
#pragma unroll
  for (int j = 0; j < 8; ++j) inv[j] = 1.0f / lrow[j];
#pragma unroll
  for (int nt = 0; nt < 8; ++nt)
#pragma unroll
    for (int j = 0; j < 8; ++j) {
      int row = t0 + j + 8 * hi;
      int col = nt * 16 + c;
      ah[((size_t)(b * SEQ_ + row) * HQ_ + h) * HD_ + col] =
          (_Float16)(o[nt][j] * inv[j]);
    }
}

// ---------------------------------------------------------------------------
// Host-side launch
// ---------------------------------------------------------------------------
extern "C" void kernel_launch(void* const* d_in, const int* in_sizes, int n_in,
                              void* d_out, int out_size, void* d_ws, size_t ws_size,
                              hipStream_t stream) {
  const float* x  = (const float*)d_in[0];
  const float* wq = (const float*)d_in[1];
  const float* wk = (const float*)d_in[2];
  const float* wv = (const float*)d_in[3];
  const float* wo = (const float*)d_in[4];
  float* out = (float*)d_out;

  const size_t M = BATCH_ * SEQ_;          // 4096 rows
  const size_t D = DMODEL_;                // 2048
  const size_t NQ = HQ_ * HD_;             // 2048
  const size_t NKV = HKV_ * HD_;           // 512

  char* ws = (char*)d_ws;
  size_t off = 0;
  auto alloc = [&](size_t halfs) {
    _Float16* p = (_Float16*)(ws + off);
    off += halfs * sizeof(_Float16);
    return p;
  };
  _Float16* xh   = alloc(M * D);
  _Float16* wqhT = alloc(D * NQ);    // [NQ][D]
  _Float16* wkhT = alloc(D * NKV);   // [NKV][D]
  _Float16* wvhT = alloc(D * NKV);   // [NKV][D]
  _Float16* wohT = alloc(NQ * D);    // [D][NQ]
  _Float16* qh   = alloc(M * NQ);
  _Float16* kh   = alloc(M * NKV);
  _Float16* vh   = alloc(M * NKV);
  _Float16* vth  = alloc(M * NKV);
  _Float16* ah   = alloc(M * NQ);
  (void)ws_size;  // ~84 MB used

  // 1) cast x; cast+transpose weights (B operands become [N][K])
  {
    int n4 = (int)(M * D / 4);
    cast_f32_f16<<<(n4 + 255) / 256, 256, 0, stream>>>(x, xh, n4);
  }
  auto castT = [&](const float* src, _Float16* dst, size_t K, size_t N) {
    int n = (int)(K * N);
    cast_t_f32_f16<<<(n + 255) / 256, 256, 0, stream>>>(src, dst, (int)K, (int)N);
  };
  castT(wq, wqhT, D, NQ);
  castT(wk, wkhT, D, NKV);
  castT(wv, wvhT, D, NKV);
  castT(wo, wohT, NQ, D);

  // 2) QKV projections (WMMA, async-LDS staging)
  wmma_gemm_kernel<_Float16><<<dim3(NQ / 128, M / 128), 256, 0, stream>>>(
      xh, wqhT, qh, (int)M, (int)NQ, (int)D);
  wmma_gemm_kernel<_Float16><<<dim3(NKV / 128, M / 128), 256, 0, stream>>>(
      xh, wkhT, kh, (int)M, (int)NKV, (int)D);
  wmma_gemm_kernel<_Float16><<<dim3(NKV / 128, M / 128), 256, 0, stream>>>(
      xh, wvhT, vh, (int)M, (int)NKV, (int)D);

  // 3) RoPE on q,k ; transpose v
  {
    int npq = BATCH_ * SEQ_ * HQ_ * (HD_ / 2);
    rope_kernel<<<(npq + 255) / 256, 256, 0, stream>>>(qh, HQ_ * 64, npq);
    int npk = BATCH_ * SEQ_ * HKV_ * (HD_ / 2);
    rope_kernel<<<(npk + 255) / 256, 256, 0, stream>>>(kh, HKV_ * 64, npk);
    int nv = BATCH_ * SEQ_ * HKV_ * HD_;
    vtrans_kernel<<<(nv + 255) / 256, 256, 0, stream>>>(vh, vth);
  }

  // 4) causal GQA flash attention (WMMA): 4096 q-tiles, 4 waves/block
  flash_attn_kernel<<<(BATCH_ * HQ_ * (SEQ_ / 16)) / 4, 128, 0, stream>>>(
      qh, kh, vth, ah);

  // 5) output projection (WMMA, f32 out)
  wmma_gemm_kernel<float><<<dim3(D / 128, M / 128), 256, 0, stream>>>(
      ah, wohT, out, (int)M, (int)D, (int)NQ);

  (void)in_sizes; (void)n_in; (void)out_size;
}
